// MultichannelMultiheadAttention_4896262717578
// MI455X (gfx1250) — matrix operand order repaired and checked by reading, not compiled, not run on hardware
//
#include <hip/hip_runtime.h>

#define CCH 8
#define FF  256
#define NHD 8
#define DH  32
#define WW  1024

typedef __attribute__((ext_vector_type(8)))  float          v8f;
typedef __attribute__((ext_vector_type(16))) __bf16         v16bf;
typedef __attribute__((ext_vector_type(16))) unsigned short v16u;

union Frag2 { v16u v; uint4 q[2]; };          // assemble 32B fragment from two 16B loads
union Pack8 { uint4 q; unsigned short s[8]; };

__device__ __forceinline__ unsigned short f2bf(float f) {
    unsigned int u = __float_as_uint(f);
    u += 0x7FFFu + ((u >> 16) & 1u);          // round-to-nearest-even
    return (unsigned short)(u >> 16);
}

__device__ __forceinline__ v8f wmma_bf16(v16u a, v16u b, v8f c) {
    return __builtin_amdgcn_wmma_f32_16x16x32_bf16(
        false, __builtin_bit_cast(v16bf, a),
        false, __builtin_bit_cast(v16bf, b),
        (short)0, c, false, false);
}

// ---------------------------------------------------------------- conv 3x3
// idx bits: p(2) c(3) w(10) f(8); output transposed bf16 [p][c][w][f]
__global__ __launch_bounds__(256)
void conv3x3_kernel(const float* __restrict__ x,
                    const float* __restrict__ wq,
                    const float* __restrict__ wk,
                    const float* __restrict__ wv,
                    unsigned short* __restrict__ conv_t) {
    int idx = blockIdx.x * 256 + threadIdx.x;
    int f = idx & (FF - 1);
    int w = (idx >> 8) & (WW - 1);
    int c = (idx >> 18) & 7;
    int p = idx >> 21;
    const float* wc = (p == 0) ? wq : ((p == 1) ? wk : wv);
    wc += c * CCH * 9;
    float acc = 0.f;
    for (int ci = 0; ci < CCH; ++ci) {
        const float* xp  = x + (size_t)(ci * FF) * WW;
        const float* wcc = wc + ci * 9;
        for (int df = 0; df < 3; ++df) {
            int ff = f + df - 1;
            if (ff < 0 || ff >= FF) continue;
            for (int dw = 0; dw < 3; ++dw) {
                int wp = w + dw - 1;
                if (wp < 0 || wp >= WW) continue;
                acc += xp[ff * WW + wp] * wcc[df * 3 + dw];
            }
        }
    }
    conv_t[idx] = f2bf(acc);                   // [(p*8+c)*1024 + w]*256 + f == idx
}

// ------------------------------------------- weight f32 -> bf16 conversion
// wbf layout: [slot(4)][c][g][h], slot = wq,wk,wv,wo
__global__ __launch_bounds__(256)
void wcvt_kernel(const float* __restrict__ wq_pw,
                 const float* __restrict__ wk_pw,
                 const float* __restrict__ wv_pw,
                 const float* __restrict__ wo_pw,
                 unsigned short* __restrict__ wbf) {
    int idx = blockIdx.x * 256 + threadIdx.x;  // 4 * 524288
    int slot = idx >> 19;
    int r = idx & 524287;
    const float* src = (slot == 0) ? wq_pw : (slot == 1) ? wk_pw : (slot == 2) ? wv_pw : wo_pw;
    wbf[idx] = f2bf(src[r]);
}

// ------------------------------------------------- per-channel F x F GEMM
// C[g][w] = sum_h A[g][h] * Bt[w][h]  (A bf16 256x256, Bt bf16 1024x256)
// mode 0: z=proj*8+c; proj 0/1 -> q/k head-major [pc][n][w][d] (packed b128
//         stores); proj 2 -> v transposed [c*8+n][d][w]
// mode 1: z=c; f32 out [c][g][w]
__global__ __launch_bounds__(256)
void pw_gemm_kernel(const unsigned short* __restrict__ wbf,
                    const unsigned short* __restrict__ conv_t,
                    unsigned short* __restrict__ qkv_bf,
                    const unsigned short* __restrict__ o_t,
                    float* __restrict__ out_f32,
                    int mode) {
    const int tid  = threadIdx.x;
    const int wave = tid >> 5, lane = tid & 31;
    const int hi   = lane >> 4, ln  = lane & 15;
    const int g0 = blockIdx.y * 32 + (wave >> 2) * 16;
    const int w0 = blockIdx.x * 256 + (wave & 3) * 64;
    const unsigned short* A;
    const unsigned short* Bt;
    int proj = 0, c;
    if (mode == 0) {
        proj = blockIdx.z >> 3; c = blockIdx.z & 7;
        A  = wbf + ((size_t)proj * CCH + c) * FF * FF;
        Bt = conv_t + ((size_t)(proj * CCH + c)) * WW * FF;
    } else {
        c  = blockIdx.z;
        A  = wbf + ((size_t)3 * CCH + c) * FF * FF;
        Bt = o_t + (size_t)c * WW * FF;
    }
    v8f acc[4] = {};
    for (int h0 = 0; h0 < FF; h0 += 32) {
        Frag2 au;
        const unsigned short* ap = A + (g0 + ln) * FF + h0 + hi * 8;
        au.q[0] = *(const uint4*)(ap);
        au.q[1] = *(const uint4*)(ap + 16);
#pragma unroll
        for (int t = 0; t < 4; ++t) {
            Frag2 bf; const unsigned short* bp = Bt + (size_t)(w0 + t * 16 + ln) * FF + h0 + hi * 8;
            bf.q[0] = *(const uint4*)(bp); bf.q[1] = *(const uint4*)(bp + 16);
            v16u bu = bf.v;
            acc[t] = wmma_bf16(au.v, bu, acc[t]);
        }
    }
    if (mode == 1) {
#pragma unroll
        for (int t = 0; t < 4; ++t)
#pragma unroll
            for (int r = 0; r < 8; ++r)
                out_f32[((size_t)c * FF + g0 + r + hi * 8) * WW + w0 + t * 16 + ln] = acc[t][r];
    } else if (proj < 2) {
        // q/k head-major: lane holds 8 consecutive d values per tile -> b128 store
        int nh = g0 >> 5;
        int dbase = (g0 & 31) + hi * 8;
        size_t base = ((size_t)(proj * CCH + c) * NHD + nh) * WW;
#pragma unroll
        for (int t = 0; t < 4; ++t) {
            Pack8 pk;
#pragma unroll
            for (int r = 0; r < 8; ++r) pk.s[r] = f2bf(acc[t][r]);
            int w = w0 + t * 16 + ln;
            *(uint4*)(qkv_bf + (base + w) * DH + dbase) = pk.q;
        }
    } else {
        // v transposed [c*8+nh][d][w], region offset 2*2097152 elements
#pragma unroll
        for (int t = 0; t < 4; ++t)
#pragma unroll
            for (int r = 0; r < 8; ++r) {
                int g = g0 + r + hi * 8;
                int nh = g >> 5, dd = g & 31;
                qkv_bf[(size_t)4194304 + ((size_t)(c * NHD + nh) * DH + dd) * WW
                       + w0 + t * 16 + ln] = f2bf(acc[t][r]);
            }
    }
}

// ------------------------------------------------------- fused attention
// grid: (64 q-tiles, 64 c*n).  8 waves, each owns 128 k-columns.
__global__ __launch_bounds__(256)
void attn_kernel(const unsigned short* __restrict__ qkv_bf,
                 const float* __restrict__ prev_qk,
                 float* __restrict__ qk_out,
                 float* __restrict__ a_t) {
    __shared__ unsigned short P_lds[16 * WW];      // 32 KB softmax probs (bf16)
    __shared__ float red[8 * 16 * DH];             // 16 KB per-wave AV partials
    __shared__ float wred[8 * 16];                 // per-wave row stats

    const int tid  = threadIdx.x;
    const int wave = tid >> 5, lane = tid & 31;
    const int hi   = lane >> 4, ln  = lane & 15;
    const int cn = blockIdx.y;                     // c*8 + n
    const int q0 = blockIdx.x * 16;
    const size_t qb = (size_t)cn * WW * DH;                 // q head-major
    const size_t kb = (size_t)2097152 + qb;                 // k head-major
    const size_t vb = (size_t)4194304 + (size_t)cn * DH * WW; // v transposed [d][w]
    const int kw0 = wave * 128;

    // prefetch the dominant stream (prev_qk rows for this tile)
    __builtin_prefetch(prev_qk + ((size_t)cn * WW + q0 + ln) * WW + kw0, 0, 1);

    // Q fragment (A): rows q0..q0+15, two b128 loads
    Frag2 aq;
    {
        const unsigned short* qp = qkv_bf + qb + (size_t)(q0 + ln) * DH + hi * 8;
        aq.q[0] = *(const uint4*)(qp);
        aq.q[1] = *(const uint4*)(qp + 16);
    }

    // S = Q K^T over this wave's 128 cols: 8 WMMAs, dh=32 = single K-step
    v8f acc[8];
#pragma unroll
    for (int t = 0; t < 8; ++t) {
        int kcol = kw0 + t * 16 + ln;
        Frag2 bf; const unsigned short* bp = qkv_bf + kb + (size_t)kcol * DH + hi * 8;
        bf.q[0] = *(const uint4*)(bp); bf.q[1] = *(const uint4*)(bp + 16);
        v16u bk = bf.v;
        v8f z = {};
        acc[t] = wmma_bf16(aq.v, bk, z);
    }

    // scale + prev_qk, single-pass store of qk output
    const float scale = 0.0625f;                   // 1/sqrt(256)
#pragma unroll
    for (int t = 0; t < 8; ++t) {
#pragma unroll
        for (int r = 0; r < 8; ++r) {
            int m = r + hi * 8;
            int col = kw0 + t * 16 + ln;
            size_t gi = ((size_t)cn * WW + (q0 + m)) * WW + col;
            float v = acc[t][r] * scale + prev_qk[gi];
            qk_out[gi] = v;
            acc[t][r] = v;
        }
    }

    // row max: per-lane over 8 tiles, half-wave shuffle, cross-wave via LDS
    float vm[8];
#pragma unroll
    for (int r = 0; r < 8; ++r) {
        float m = acc[0][r];
#pragma unroll
        for (int t = 1; t < 8; ++t) m = fmaxf(m, acc[t][r]);
#pragma unroll
        for (int off = 1; off < 16; off <<= 1) m = fmaxf(m, __shfl_xor(m, off, 32));
        vm[r] = m;
    }
    if (ln == 0) {
#pragma unroll
        for (int r = 0; r < 8; ++r) wred[wave * 16 + r + hi * 8] = vm[r];
    }
    __syncthreads();
    float rmax[8];
#pragma unroll
    for (int r = 0; r < 8; ++r) {
        float m = wred[r + hi * 8];
        for (int w2 = 1; w2 < 8; ++w2) m = fmaxf(m, wred[w2 * 16 + r + hi * 8]);
        rmax[r] = m;
    }
    __syncthreads();

    // exp + row sum
    float vs[8];
#pragma unroll
    for (int r = 0; r < 8; ++r) {
        float s = 0.f;
#pragma unroll
        for (int t = 0; t < 8; ++t) {
            float e = __expf(acc[t][r] - rmax[r]);
            acc[t][r] = e;
            s += e;
        }
#pragma unroll
        for (int off = 1; off < 16; off <<= 1) s += __shfl_xor(s, off, 32);
        vs[r] = s;
    }
    if (ln == 0) {
#pragma unroll
        for (int r = 0; r < 8; ++r) wred[wave * 16 + r + hi * 8] = vs[r];
    }
    __syncthreads();
    float rinv[8];
#pragma unroll
    for (int r = 0; r < 8; ++r) {
        float s = 0.f;
        for (int w2 = 0; w2 < 8; ++w2) s += wred[w2 * 16 + r + hi * 8];
        rinv[r] = 1.0f / s;
    }

    // normalized probs -> LDS as bf16 (A-fragment source for AV)
#pragma unroll
    for (int t = 0; t < 8; ++t) {
#pragma unroll
        for (int r = 0; r < 8; ++r) {
            int m = r + hi * 8;
            int col = kw0 + t * 16 + ln;
            P_lds[m * WW + col] = f2bf(acc[t][r] * rinv[r]);
        }
    }
    __syncthreads();

    // AV partial over this wave's 128 k-cols: 4 K-chunks x 2 d-tiles
    v8f av[2] = {};
#pragma unroll
    for (int kc = 0; kc < 4; ++kc) {
        Frag2 ap;
        const unsigned short* pp = P_lds + ln * WW + kw0 + kc * 32 + hi * 8;
        ap.q[0] = *(const uint4*)(pp);             // ds_load_b128
        ap.q[1] = *(const uint4*)(pp + 16);
#pragma unroll
        for (int dt = 0; dt < 2; ++dt) {
            Frag2 bf; const unsigned short* bp = qkv_bf + vb + (size_t)(dt * 16 + ln) * WW
                                     + kw0 + kc * 32 + hi * 8;
            bf.q[0] = *(const uint4*)(bp); bf.q[1] = *(const uint4*)(bp + 16);
            v16u bv = bf.v;
            av[dt] = wmma_bf16(ap.v, bv, av[dt]);
        }
    }

    // deterministic cross-wave reduction of 16x32 partials
#pragma unroll
    for (int dt = 0; dt < 2; ++dt) {
#pragma unroll
        for (int r = 0; r < 8; ++r) {
            int m = r + hi * 8;
            int d = dt * 16 + ln;
            red[(wave * 16 + m) * DH + d] = av[dt][r];
        }
    }
    __syncthreads();
#pragma unroll
    for (int j = 0; j < 2; ++j) {
        int idx = tid * 2 + j;
        int m = idx >> 5;
        int d = idx & 31;
        float s = 0.f;
        for (int w2 = 0; w2 < 8; ++w2) s += red[(w2 * 16 + m) * DH + d];
        int c = cn >> 3, n = cn & 7;
        a_t[((size_t)c * WW + q0 + m) * FF + n * DH + d] = s;  // a transposed [c][w][h]
    }
}

// ------------------------------------------------ depthwise channel mix
// reads a_t [c][w][h] f32 (coalesced), writes o_t [d][w][h] bf16 (coalesced)
__global__ __launch_bounds__(256)
void dwmix_kernel(const float* __restrict__ a_t,
                  const float* __restrict__ wo_dw,
                  unsigned short* __restrict__ o_t) {
    int idx = blockIdx.x * 256 + threadIdx.x;      // d(3) w(10) f(8)
    int f = idx & 255;
    int w = (idx >> 8) & 1023;
    int d = idx >> 18;
    float acc = 0.f;
#pragma unroll
    for (int c = 0; c < 8; ++c)
        acc += wo_dw[d * 8 + c] * a_t[((size_t)c * WW + w) * FF + f];
    o_t[idx] = f2bf(acc);                          // [(d*1024+w)*256+f] == idx
}

extern "C" void kernel_launch(void* const* d_in, const int* in_sizes, int n_in,
                              void* d_out, int out_size, void* d_ws, size_t ws_size,
                              hipStream_t stream) {
    const float* x       = (const float*)d_in[0];
    const float* prev_qk = (const float*)d_in[1];
    const float* wq_conv = (const float*)d_in[2];
    const float* wk_conv = (const float*)d_in[3];
    const float* wv_conv = (const float*)d_in[4];
    const float* wq_pw   = (const float*)d_in[5];
    const float* wk_pw   = (const float*)d_in[6];
    const float* wv_pw   = (const float*)d_in[7];
    const float* wo_dw   = (const float*)d_in[8];
    const float* wo_pw   = (const float*)d_in[9];

    float* out    = (float*)d_out;                     // [8][256][1024]
    float* qk_out = out + (size_t)CCH * FF * WW;       // [8][8][1024][1024]

    // workspace (~40 MB):
    //   conv_t  bf16  [3][8][1024][256]   6291456 u16
    //   qkv_bf  bf16  q/k head-major + v transposed   6291456 u16
    //   wbf     bf16  [4][8][256][256]    2097152 u16
    //   o_t     bf16  [8][1024][256]      2097152 u16
    //   a_t     f32   [8][1024][256]      2097152 f32
    unsigned short* conv_t = (unsigned short*)d_ws;
    unsigned short* qkv_bf = conv_t + 6291456;
    unsigned short* wbf    = qkv_bf + 6291456;
    unsigned short* o_t    = wbf + 2097152;
    float*          a_t    = (float*)(o_t + 2097152);

    conv3x3_kernel<<<dim3(6291456 / 256), 256, 0, stream>>>(x, wq_conv, wk_conv, wv_conv, conv_t);
    wcvt_kernel<<<dim3(2097152 / 256), 256, 0, stream>>>(wq_pw, wk_pw, wv_pw, wo_pw, wbf);
    pw_gemm_kernel<<<dim3(4, 8, 24), 256, 0, stream>>>(wbf, conv_t, qkv_bf, o_t, out, 0);
    attn_kernel<<<dim3(64, 64), 256, 0, stream>>>(qkv_bf, prev_qk, qk_out, a_t);
    dwmix_kernel<<<dim3(2097152 / 256), 256, 0, stream>>>(a_t, wo_dw, o_t);
    pw_gemm_kernel<<<dim3(4, 8, 8), 256, 0, stream>>>(wbf, conv_t, qkv_bf, o_t, out, 1);
}
